// VectorQuantizer_51556787421368
// MI455X (gfx1250) — matrix operand order repaired and checked by reading, not compiled, not run on hardware
//
#include <hip/hip_runtime.h>

// ---------------------------------------------------------------------------
// VQ-VAE vector quantizer for MI455X (gfx1250, wave32, WMMA).
//
// z   : (8, 64, 16, 64, 64) f32   -> per batch a (D=64 x S=65536) row-major
//                                    matrix, i.e. already the WMMA B operand
//                                    (K-dim = channels, N = spatial cols).
// emb : (512, 64) f32 codebook.
//
// dist argmin uses  score_k(n) = ||e_k||^2 - 2 * e_k . z(:,n)
// computed as a bf16 WMMA GEMM:  A = (-2*emb) bf16 (512x64), B = z bf16 tile.
// Output value of the straight-through estimator == z_q numerically, and
// loss == 1.25 * mean((z_q - z)^2).
// ---------------------------------------------------------------------------

typedef __attribute__((ext_vector_type(8)))  float  v8f;
typedef __attribute__((ext_vector_type(16))) __bf16 v16bf;

#define KC    512         // codes
#define DC    64          // channels
#define SP    65536       // spatial per batch (16*64*64)
#define NB    8           // batches
#define COLS  128         // spatial columns per workgroup
#define NWG   4096        // NB*SP/COLS
#define ZQN   33554432ull // NB*DC*SP  (z_q element count)

__device__ __forceinline__ unsigned short f2bf(float x) {
  __bf16 h = (__bf16)x;
  return __builtin_bit_cast(unsigned short, h);
}

// ---------------------------------------------------------------------------
// Pack kernel: emb -> (-2*emb) bf16 in WMMA A-fragment order, plus ||e_k||^2.
// A-fragment (16-bit A, 16x32, ISA 7.12.2): lane<16 holds M=lane,
//   VGPR r pair = K in {2r,2r+1} (r<4) or {16+2(r-4), ...}; lanes>=16: +8 on K.
// Layout: wsA[((t*2+j)*32 + lane)*16 + s], t=code tile (0..31), j=K-step.
// ---------------------------------------------------------------------------
__global__ __launch_bounds__(256)
void vq_pack(const float* __restrict__ emb,
             unsigned short* __restrict__ wsA,
             float* __restrict__ e2) {
  int id = blockIdx.x * 256 + threadIdx.x;        // 128 blocks * 256 = 32768
  if (id < 32 * 2 * 32 * 16) {
    int s    = id & 15;
    int lane = (id >> 4) & 31;
    int tj   = id >> 9;
    int j    = tj & 1, t = tj >> 1;
    int half = lane >> 4, m = lane & 15;
    int r    = s >> 1, pos = s & 1;
    int kk   = ((r < 4) ? (2 * r) : (16 + 2 * (r - 4))) + 8 * half + pos;
    int c    = 32 * j + kk;                       // channel
    int code = 16 * t + m;
    wsA[id] = f2bf(-2.0f * emb[code * DC + c]);
  }
  if (id < KC) {
    float s2 = 0.f;
    #pragma unroll
    for (int c = 0; c < DC; ++c) { float v = emb[id * DC + c]; s2 += v * v; }
    e2[id] = s2;
  }
}

// ---------------------------------------------------------------------------
// Main kernel: 256 threads = 8 waves; wave w owns 16 spatial columns.
// Phase 1: bf16 WMMA distance GEMM + per-lane running argmin over 512 codes.
// Phase 2: gather emb[idx] (f32), write z_q, accumulate squared error.
// ---------------------------------------------------------------------------
__global__ __launch_bounds__(256)
void vq_main(const float* __restrict__ z,
             const float* __restrict__ emb,
             const unsigned short* __restrict__ wsAv,
             const float* __restrict__ e2g,
             float* __restrict__ out,        // [z_q (ZQN)] [loss (1)] [indices (NB*SP) as float]
             float* __restrict__ partials) { // per-WG loss partial sums
  __shared__ float e2_s[KC];
  __shared__ int   idx_s[COLS];
  __shared__ float red_s[8];

  const int tid  = threadIdx.x;
  const int wg   = blockIdx.x;
  const int b    = wg >> 9;                  // 512 WGs per batch
  const int sp0  = (wg & 511) << 7;          // column base within batch
  const int lane = tid & 31;
  const int wave = tid >> 5;
  const int half = lane >> 4;
  const int col  = lane & 15;

  for (int i = tid; i < KC; i += 256) e2_s[i] = e2g[i];

  const float* zb = z + (size_t)b * DC * SP;
  const int sp = sp0 + wave * 16 + col;      // this lane's spatial column

  // B fragments (16-bit B 32x16): lane holds column sp, K = 8*half + s (s<8),
  // 16 + 8*half + (s-8) (s>=8) -> 16 strided f32 loads + cvt each.
  v16bf B0, B1;
  {
    const int c0 = 8 * half;
    #pragma unroll
    for (int s = 0; s < 16; ++s)
      B0[s] = (__bf16)zb[(size_t)(c0 + s + (s >= 8 ? 8 : 0)) * SP + sp];
    #pragma unroll
    for (int s = 0; s < 16; ++s)
      B1[s] = (__bf16)zb[(size_t)(32 + c0 + s + (s >= 8 ? 8 : 0)) * SP + sp];
  }
  __syncthreads();                            // e2_s ready

  const v16bf* A = (const v16bf*)wsAv;        // 32-byte aligned fragments
  float best = 3.4028235e38f;
  int   bidx = 0;

  for (int t = 0; t < 32; ++t) {              // 32 code tiles of 16
    v16bf a0 = A[(t * 2 + 0) * 32 + lane];
    v16bf a1 = A[(t * 2 + 1) * 32 + lane];
    v8f acc = {};
    acc = __builtin_amdgcn_wmma_f32_16x16x32_bf16(false, a0, false, B0,
                                                  (short)0, acc, false, false);
    acc = __builtin_amdgcn_wmma_f32_16x16x32_bf16(false, a1, false, B1,
                                                  (short)0, acc, false, false);
    // C layout: VGPR r -> code = 16t + r + 8*half, column = sp (per lane).
    const int cb = t * 16 + half * 8;
    #pragma unroll
    for (int r = 0; r < 8; ++r) {
      float sc = acc[r] + e2_s[cb + r];
      bool lt = sc < best;
      best = lt ? sc : best;
      bidx = lt ? (cb + r) : bidx;
    }
  }

  // lanes l and l^16 cover the same column with disjoint code halves
  {
    float ob = __shfl_xor(best, 16, 32);
    int   oi = __shfl_xor(bidx, 16, 32);
    bool take = (ob < best) || (ob == best && oi < bidx); // first-index ties
    best = take ? ob : best;
    bidx = take ? oi : bidx;
  }
  if (half == 0) idx_s[wave * 16 + col] = bidx;
  __syncthreads();

  // ---- Phase 2: gather + output + loss -----------------------------------
  const int nl    = tid & 127;                // column within WG tile
  const int c0    = (tid >> 7) * 32;          // channel half
  const int spg   = sp0 + nl;
  const int myidx = idx_s[nl];
  const float* eb = emb + myidx * DC;

  float lacc = 0.f;
  const size_t base = (size_t)b * DC * SP + (size_t)spg;
  #pragma unroll 8
  for (int c = c0; c < c0 + 32; ++c) {
    size_t off = base + (size_t)c * SP;
    float zv = z[off];
    float qv = eb[c];
    out[off] = qv;                            // z_q_st == z_q numerically
    float d = qv - zv;
    lacc += d * d;
  }
  if (tid < COLS)
    out[ZQN + 1 + (size_t)b * SP + spg] = (float)myidx;   // indices output

  // deterministic in-block reduction
  #pragma unroll
  for (int o = 16; o > 0; o >>= 1) lacc += __shfl_xor(lacc, o, 32);
  if (lane == 0) red_s[wave] = lacc;
  __syncthreads();
  if (tid == 0) {
    float s = 0.f;
    #pragma unroll
    for (int w = 0; w < 8; ++w) s += red_s[w];
    partials[wg] = s;
  }
}

// Deterministic final loss reduction (fixed order, no float atomics).
__global__ __launch_bounds__(256)
void vq_loss(const float* __restrict__ partials, float* __restrict__ loss_out) {
  __shared__ float sm[256];
  float a = 0.f;
  for (int i = threadIdx.x; i < NWG; i += 256) a += partials[i];
  sm[threadIdx.x] = a;
  __syncthreads();
  for (int s = 128; s > 0; s >>= 1) {
    if (threadIdx.x < s) sm[threadIdx.x] += sm[threadIdx.x + s];
    __syncthreads();
  }
  if (threadIdx.x == 0)
    loss_out[0] = sm[0] * (1.25f / (float)ZQN);   // (1+beta) * mean
}

extern "C" void kernel_launch(void* const* d_in, const int* in_sizes, int n_in,
                              void* d_out, int out_size, void* d_ws, size_t ws_size,
                              hipStream_t stream) {
  const float* z   = (const float*)d_in[0];   // (8,64,16,64,64) f32
  const float* emb = (const float*)d_in[1];   // (512,64) f32
  float* out = (float*)d_out;

  // workspace layout: [ packed A bf16 : 64KB ][ e2 : 2KB ][ partials : 16KB ]
  char* ws = (char*)d_ws;
  unsigned short* wsA = (unsigned short*)ws;
  float* e2           = (float*)(ws + 65536);
  float* partials     = (float*)(ws + 65536 + 2048);

  hipLaunchKernelGGL(vq_pack, dim3(128), dim3(256), 0, stream, emb, wsA, e2);
  hipLaunchKernelGGL(vq_main, dim3(NWG), dim3(256), 0, stream,
                     z, emb, wsA, e2, out, partials);
  hipLaunchKernelGGL(vq_loss, dim3(1), dim3(256), 0, stream,
                     partials, out + ZQN);
}
